// RNNModel_28415503630691
// MI455X (gfx1250) — matrix operand order repaired and checked by reading, not compiled, not run on hardware
//
#include <hip/hip_runtime.h>

#define B_SZ   512
#define T_SZ   1000
#define IN_SZ  39
#define H_SZ   64
#define OUT_SZ 48
#define HSTR   72           // f16 stride for h buffer: 144B = 36 dwords -> bank-skewed rows
#define XSTR   72           // f16 stride for x buffers (39 data + zeros to 64 + pad)

typedef _Float16 f16;
typedef __attribute__((ext_vector_type(16))) _Float16 v16h;
typedef __attribute__((ext_vector_type(8)))  _Float16 v8h;
typedef __attribute__((ext_vector_type(8)))  float    v8f;

__device__ __forceinline__ float frcp_f(float x)  { return __builtin_amdgcn_rcpf(x); }
__device__ __forceinline__ float fexp2_f(float x) { return __builtin_amdgcn_exp2f(x); }
// sigmoid/tanh with correct +-inf limits, no clamps needed
__device__ __forceinline__ float fsigmoid(float x) {
    return frcp_f(1.0f + fexp2_f(-1.44269504f * x));
}
__device__ __forceinline__ float ftanh_f(float x) {
    float e = fexp2_f(2.88539008f * x);
    return 1.0f - 2.0f * frcp_f(e + 1.0f);
}

__device__ __forceinline__ v8f splat8(float x) {
    v8f r;
#pragma unroll
    for (int i = 0; i < 8; ++i) r[i] = x;
    return r;
}

// A fragment (16x32 f16), stride-72 buffer. lane m = L%16, hi = L>>4:
// halves 0..7  = buf[m][k32 + hi*16 + 0..7], halves 8..15 = buf[m][k32 + hi*16 + 8..15]
__device__ __forceinline__ v16h load_frag72(const f16* buf, int m, int hi, int k32) {
    const f16* p = buf + m * HSTR + k32 + hi * 16;
    v8h lo = *(const v8h*)(p);
    v8h hv = *(const v8h*)(p + 8);
    v16h a;
#pragma unroll
    for (int i = 0; i < 8; ++i) { a[i] = lo[i]; a[i + 8] = hv[i]; }
    return a;
}

__launch_bounds__(256, 1)
__global__ void lstm_persistent_kernel(const float* __restrict__ x,
                                       const float* __restrict__ W_ih,
                                       const float* __restrict__ W_hh,
                                       const float* __restrict__ b_ih,
                                       const float* __restrict__ b_hh,
                                       const float* __restrict__ W_out,
                                       const float* __restrict__ b_out,
                                       float* __restrict__ out)
{
    __shared__ f16 Hlds[16 * HSTR];       // h_t, f16, rows = batch rows
    __shared__ f16 Xlds[2][16 * XSTR];    // double-buffered x_t (cols 39..63 stay zero)

    const int tid    = threadIdx.x;
    const int wave   = tid >> 5;          // 0..7 (wave32)
    const int lane   = tid & 31;
    const int lane16 = lane & 15;
    const int hi     = (lane >> 4) & 1;
    const int wgbase = blockIdx.x * 16;

    // ---- zero h (h0 = 0) and both x buffers (zero-pad region) ----
    for (int i = tid; i < 16 * HSTR; i += 256) Hlds[i] = (f16)0.0f;
    for (int i = tid; i < 2 * 16 * XSTR; i += 256) (&Xlds[0][0])[i] = (f16)0.0f;

    // ================= role: gate waves (0..3) =================
    // wave g owns h-columns j = g*16 + lane16; its 4 N-tiles are gates {i,f,g,o}
    // at column n_global = q*64 + g*16 + lane16.
    // B fragment: half p -> K = kf*32 + hi*16 + p;  Wc[k][n] = (k<64) W_hh[n][k]
    //                                               (k<103) W_ih[n][k-64] else 0.
    v16h bg[4][4];
    float biasq[4];
    if (wave < 4) {
#pragma unroll
        for (int q = 0; q < 4; ++q) {
            const int n = q * 64 + wave * 16 + lane16;
            biasq[q] = b_ih[n] + b_hh[n];
#pragma unroll
            for (int kf = 0; kf < 4; ++kf) {
                v16h f;
#pragma unroll
                for (int p = 0; p < 16; ++p) {
                    const int k = kf * 32 + hi * 16 + p;
                    float w;
                    if (k < H_SZ)              w = W_hh[n * H_SZ + k];
                    else if (k < H_SZ + IN_SZ) w = W_ih[n * IN_SZ + (k - H_SZ)];
                    else                       w = 0.0f;
                    f[p] = (f16)w;
                }
                bg[q][kf] = f;
            }
        }
    }

    // ================= role: output waves (4..6) =================
    v16h by[2];
    float ybias = 0.0f;
    const int ny = ((wave >= 4 && wave < 7) ? (wave - 4) : 0) * 16 + lane16;  // 0..47
    if (wave >= 4) {
        ybias = b_out[ny];
#pragma unroll
        for (int kf = 0; kf < 2; ++kf) {
            v16h f;
#pragma unroll
            for (int p = 0; p < 16; ++p) {
                const int k = kf * 32 + hi * 16 + p;
                f[p] = (f16)W_out[ny * H_SZ + k];
            }
            by[kf] = f;
        }
    }

    __syncthreads();   // zeros visible

    // ---- x loader role: waves 4..7, lane u = 0..127 covers 16 rows x 39 cols ----
    const int u  = tid - 128;             // valid when wave >= 4
    const int xr = u >> 3;                // 0..15
    const int xi = (u & 7) * 5;           // 0,5,...,35
    if (wave >= 4) {                      // load x_0 into buffer 0
        const size_t xb = ((size_t)(wgbase + xr) * T_SZ) * IN_SZ;
#pragma unroll
        for (int q = 0; q < 5; ++q) {
            const int ii = xi + q;
            if (ii < IN_SZ) Xlds[0][xr * XSTR + ii] = (f16)x[xb + ii];
        }
    }

    float cst[8];                          // cell state: rows hi*8..hi*8+7 at col j
#pragma unroll
    for (int i = 0; i < 8; ++i) cst[i] = 0.0f;
    f16 hreg[8];

#pragma unroll 1
    for (int t = 0; t < T_SZ; ++t) {
        __syncthreads();   // (A) Hlds = h_{t-1}, Xlds[t&1] = x_t

        if (wave < 4) {
            // ---- gates: 16 x v_wmma_f32_16x16x32_f16, 4 independent chains ----
            const f16* Xp = &Xlds[t & 1][0];
            v16h a0 = load_frag72(Hlds, lane16, hi, 0);
            v16h a1 = load_frag72(Hlds, lane16, hi, 32);
            v16h a2 = load_frag72(Xp,   lane16, hi, 0);
            v16h a3 = load_frag72(Xp,   lane16, hi, 32);
            v8f ai = splat8(biasq[0]);
            v8f af = splat8(biasq[1]);
            v8f ag = splat8(biasq[2]);
            v8f ao = splat8(biasq[3]);
            ai = __builtin_amdgcn_wmma_f32_16x16x32_f16(false, a0, false, bg[0][0], (short)0, ai, false, false);
            af = __builtin_amdgcn_wmma_f32_16x16x32_f16(false, a0, false, bg[1][0], (short)0, af, false, false);
            ag = __builtin_amdgcn_wmma_f32_16x16x32_f16(false, a0, false, bg[2][0], (short)0, ag, false, false);
            ao = __builtin_amdgcn_wmma_f32_16x16x32_f16(false, a0, false, bg[3][0], (short)0, ao, false, false);
            ai = __builtin_amdgcn_wmma_f32_16x16x32_f16(false, a1, false, bg[0][1], (short)0, ai, false, false);
            af = __builtin_amdgcn_wmma_f32_16x16x32_f16(false, a1, false, bg[1][1], (short)0, af, false, false);
            ag = __builtin_amdgcn_wmma_f32_16x16x32_f16(false, a1, false, bg[2][1], (short)0, ag, false, false);
            ao = __builtin_amdgcn_wmma_f32_16x16x32_f16(false, a1, false, bg[3][1], (short)0, ao, false, false);
            ai = __builtin_amdgcn_wmma_f32_16x16x32_f16(false, a2, false, bg[0][2], (short)0, ai, false, false);
            af = __builtin_amdgcn_wmma_f32_16x16x32_f16(false, a2, false, bg[1][2], (short)0, af, false, false);
            ag = __builtin_amdgcn_wmma_f32_16x16x32_f16(false, a2, false, bg[2][2], (short)0, ag, false, false);
            ao = __builtin_amdgcn_wmma_f32_16x16x32_f16(false, a2, false, bg[3][2], (short)0, ao, false, false);
            ai = __builtin_amdgcn_wmma_f32_16x16x32_f16(false, a3, false, bg[0][3], (short)0, ai, false, false);
            af = __builtin_amdgcn_wmma_f32_16x16x32_f16(false, a3, false, bg[1][3], (short)0, af, false, false);
            ag = __builtin_amdgcn_wmma_f32_16x16x32_f16(false, a3, false, bg[2][3], (short)0, ag, false, false);
            ao = __builtin_amdgcn_wmma_f32_16x16x32_f16(false, a3, false, bg[3][3], (short)0, ao, false, false);

            // ---- LSTM update entirely in registers ----
#pragma unroll
            for (int i = 0; i < 8; ++i) {
                const float iv = fsigmoid(ai[i]);
                const float fv = fsigmoid(af[i]);
                const float gv = ftanh_f(ag[i]);
                const float ov = fsigmoid(ao[i]);
                const float c  = fv * cst[i] + iv * gv;
                cst[i] = c;
                hreg[i] = (f16)(ov * ftanh_f(c));
            }
        } else {
            // ---- y_{t-1} = sigmoid(h_{t-1} W_out^T + b_out), waves 4..6 ----
            if (wave < 7 && t > 0) {
                v8f acc = splat8(ybias);
                v16h a0 = load_frag72(Hlds, lane16, hi, 0);
                v16h a1 = load_frag72(Hlds, lane16, hi, 32);
                acc = __builtin_amdgcn_wmma_f32_16x16x32_f16(false, a0, false, by[0], (short)0, acc, false, false);
                acc = __builtin_amdgcn_wmma_f32_16x16x32_f16(false, a1, false, by[1], (short)0, acc, false, false);
#pragma unroll
                for (int i = 0; i < 8; ++i) {
                    const int m = hi * 8 + i;
                    const float yv = fsigmoid(acc[i]);
                    __builtin_nontemporal_store(
                        yv, &out[((size_t)(wgbase + m) * T_SZ + (t - 1)) * OUT_SZ + ny]);
                }
            }
            // ---- stream x_{t+1} into the other buffer (waves 4..7) ----
            if (t + 1 < T_SZ) {
                f16* Xn = &Xlds[(t + 1) & 1][0];
                const size_t xb = ((size_t)(wgbase + xr) * T_SZ + (t + 1)) * IN_SZ;
#pragma unroll
                for (int q = 0; q < 5; ++q) {
                    const int ii = xi + q;
                    if (ii < IN_SZ) Xn[xr * XSTR + ii] = (f16)__builtin_nontemporal_load(&x[xb + ii]);
                }
            }
        }

        __syncthreads();   // (B) all reads of h_{t-1} complete

        if (wave < 4) {    // publish h_t
            const int j = wave * 16 + lane16;
#pragma unroll
            for (int i = 0; i < 8; ++i) Hlds[(hi * 8 + i) * HSTR + j] = hreg[i];
        }
    }

    // ---- drain the pipeline: y_{T-1} ----
    __syncthreads();
    if (wave >= 4 && wave < 7) {
        v8f acc = splat8(ybias);
        v16h a0 = load_frag72(Hlds, lane16, hi, 0);
        v16h a1 = load_frag72(Hlds, lane16, hi, 32);
        acc = __builtin_amdgcn_wmma_f32_16x16x32_f16(false, a0, false, by[0], (short)0, acc, false, false);
        acc = __builtin_amdgcn_wmma_f32_16x16x32_f16(false, a1, false, by[1], (short)0, acc, false, false);
#pragma unroll
        for (int i = 0; i < 8; ++i) {
            const int m = hi * 8 + i;
            const float yv = fsigmoid(acc[i]);
            __builtin_nontemporal_store(
                yv, &out[((size_t)(wgbase + m) * T_SZ + (T_SZ - 1)) * OUT_SZ + ny]);
        }
    }
}

extern "C" void kernel_launch(void* const* d_in, const int* in_sizes, int n_in,
                              void* d_out, int out_size, void* d_ws, size_t ws_size,
                              hipStream_t stream) {
    (void)in_sizes; (void)n_in; (void)d_ws; (void)ws_size; (void)out_size;
    const float* x     = (const float*)d_in[0];
    const float* W_ih  = (const float*)d_in[1];
    const float* W_hh  = (const float*)d_in[2];
    const float* b_ih  = (const float*)d_in[3];
    const float* b_hh  = (const float*)d_in[4];
    const float* W_out = (const float*)d_in[5];
    const float* b_out = (const float*)d_in[6];
    float* out = (float*)d_out;

    dim3 grid(B_SZ / 16);   // 32 persistent workgroups, one 16-row batch tile each
    dim3 block(256);        // 8 wave32 waves: 4 gate waves, 3 output waves, x streamers
    lstm_persistent_kernel<<<grid, block, 0, stream>>>(
        x, W_ih, W_hh, b_ih, b_hh, W_out, b_out, out);
}
